// NeuroMotorSNN_16192026706220
// MI455X (gfx1250) — matrix operand order repaired and checked by reading, not compiled, not run on hardware
//
#include <hip/hip_runtime.h>
#include <hip/hip_fp16.h>

// ---------------------------------------------------------------------------
// NeuroMotorSNN on MI455X (gfx1250, wave32)
//
// Decomposition: grid = B/16 blocks x 8 waves. Each block owns a 16-row
// batch tile; each wave owns one 16-wide N-tile of the hidden dim.
//   - B fragments (W_in tile, f16): 4 x v16h = 32 VGPRs, loaded once.
//   - Per step: encode (v_exp), 4 x v_wmma_f32_16x16x32_f16 (K-chained
//     f32 accumulation), per-tile LN partials via __shfl_xor within
//     16-lane halves, one s_barrier with double-buffered LDS stats,
//     LIF membrane update fully in VGPRs.
// 1024 waves total, ~130 VGPRs/wave -> no VGPR-MSB churn, dense SIMDs.
// ---------------------------------------------------------------------------

typedef __attribute__((ext_vector_type(16))) _Float16 v16h;
typedef __attribute__((ext_vector_type(8)))  float    v8f;
typedef __attribute__((ext_vector_type(4)))  float    v4f;

#define HIDDEN   128
#define NTH      32
#define BETA     0.9f
#define THRESH   0.5f
#define LN_EPS   1e-5f
#define WAVES    8

__global__ __launch_bounds__(256) void snn_fused_kernel(
    const float* __restrict__ x,      // [B, T, 4]
    const float* __restrict__ W_in,   // [128, 128]
    const float* __restrict__ b_in,   // [128]
    const float* __restrict__ ln_g,   // [128]
    const float* __restrict__ ln_b,   // [128]
    const float* __restrict__ W_out,  // [2, 128]
    const float* __restrict__ b_out,  // [2]
    float* __restrict__ out,          // [B, 2]
    int B, int T)
{
    // Double-buffered per-step LN partials: (sum, sumsq) per [row][wave].
    __shared__ float2 stats[2][16][WAVES];
    // End-of-scan readout partials per [wave][row].
    __shared__ float2 robuf[WAVES][16];

    const int tid  = threadIdx.x;
    const int lane = tid & 31;         // 0..31 within wave32
    const int w    = tid >> 5;         // wave id == N-tile index (0..7)
    const int m    = lane & 15;        // A-row / B-col within 16x16 tile
    const int hs   = lane >> 4;        // half-wave select

    // ---- resident B fragments: this wave's W_in tile in f16 ---------------
    // B-matrix column n = 16*w + m; K = kc*32 + hs*8 + {0-7,16-23} per chunk.
    const int n = w * 16 + m;
    v16h bfrag[4];
#pragma unroll
    for (int kc = 0; kc < 4; ++kc) {
        const float* wp = W_in + n * HIDDEN + kc * 32 + hs * 8;
#pragma unroll
        for (int e = 0; e < 16; ++e)
            bfrag[kc][e] = (_Float16)wp[e + (e >= 8 ? 8 : 0)];
    }

    // ---- per-lane scalars for this tile (n fixed per lane) ----------------
    const float gl = ln_g[n];
    const float bl = ln_b[n];
    const float bi = b_in[n];
    const float wo0 = W_out[n];
    const float wo1 = W_out[HIDDEN + n];

    // Gaussian-encoding thresholds for this lane's 16 K-offsets of each
    // 32-wide K-chunk (16-bit A layout: low lanes K {0-7,16-23},
    // high lanes K {8-15,24-31}; one K-chunk == one input channel).
    float th[16];
#pragma unroll
    for (int h = 0; h < 16; ++h) {
        int koff = h + (h >= 8 ? 8 : 0) + hs * 8;
        th[h] = -3.0f + 6.0f * (float)koff / (float)(NTH - 1);
    }
    const float sigma  = 5.0f / (float)NTH;
    const float expcof = -0.5f / (sigma * sigma);

    // ---- recurrent state in VGPRs -----------------------------------------
    v8f mem = (v8f){};
    float ro0[8], ro1[8];
#pragma unroll
    for (int j = 0; j < 8; ++j) { ro0[j] = 0.0f; ro1[j] = 0.0f; }

    const float* xrow = x + (size_t)(blockIdx.x * 16 + m) * T * 4;

    // =======================================================================
    // Sequential scan over T timesteps (one workgroup barrier per step)
    // =======================================================================
    for (int t = 0; t < T; ++t) {
        if (t + 8 < T) __builtin_prefetch(xrow + (size_t)(t + 8) * 4, 0, 0);

        // -- encode: A fragments (16x32 f16 per K-chunk) in registers
        v4f xv = *(const v4f*)(xrow + (size_t)t * 4);
        v16h a[4];
#pragma unroll
        for (int kc = 0; kc < 4; ++kc) {
            float xc = xv[kc];
#pragma unroll
            for (int h = 0; h < 16; ++h) {
                float d = xc - th[h];
                a[kc][h] = (_Float16)__expf(d * d * expcof);
            }
        }

        // -- this tile's 16 columns of curr = spikes @ W_in^T (+ b_in)
        v8f c = (v8f){};
#pragma unroll
        for (int kc = 0; kc < 4; ++kc)
            c = __builtin_amdgcn_wmma_f32_16x16x32_f16(
                    false, a[kc], false, bfrag[kc], (short)0, c, false, false);
        v8f acc = c + bi;

        // -- per-tile LN partials: reduce across 16 lanes of each half,
        //    writer lane (m == j) stores (s, q) for row r = j + 8*hs.
        const int p = t & 1;
#pragma unroll
        for (int j = 0; j < 8; ++j) {
            float sj = acc[j];
            float qj = acc[j] * acc[j];
            sj += __shfl_xor(sj, 1);  qj += __shfl_xor(qj, 1);
            sj += __shfl_xor(sj, 2);  qj += __shfl_xor(qj, 2);
            sj += __shfl_xor(sj, 4);  qj += __shfl_xor(qj, 4);
            sj += __shfl_xor(sj, 8);  qj += __shfl_xor(qj, 8);
            if (m == j)
                stats[p][j + 8 * hs][w] = make_float2(sj, qj);
        }

        __syncthreads();   // partials of step t visible block-wide

        // -- total stats: lane handles row rr = m (upper half duplicates)
        float S = 0.0f, Q = 0.0f;
#pragma unroll
        for (int ww = 0; ww < WAVES; ++ww) {
            float2 pq = stats[p][m][ww];
            S += pq.x;  Q += pq.y;
        }
        float mu_l = S * (1.0f / HIDDEN);
        float rs_l = rsqrtf(Q * (1.0f / HIDDEN) - mu_l * mu_l + LN_EPS);

        // -- LIF membrane update + spike + readout accumulate
#pragma unroll
        for (int j = 0; j < 8; ++j) {
            // broadcast row (j + 8*hs) stats from the lane that computed them
            float mu_j = __shfl(mu_l, j + 8 * hs);
            float rs_j = __shfl(rs_l, j + 8 * hs);
            float curr = (acc[j] - mu_j) * (gl * rs_j) + bl;
            float mm   = mem[j] * BETA + curr * (1.0f - BETA);
            float spk  = (mm > THRESH) ? 1.0f : 0.0f;
            mm -= spk * THRESH;
            mem[j] = mm;
            ro0[j] += spk * wo0;
            ro1[j] += spk * wo1;
        }
    }

    // ---- final: per-tile readout sums -> LDS -> cross-wave total ----------
#pragma unroll
    for (int j = 0; j < 8; ++j) {
        float r0 = ro0[j], r1 = ro1[j];
        r0 += __shfl_xor(r0, 1);  r1 += __shfl_xor(r1, 1);
        r0 += __shfl_xor(r0, 2);  r1 += __shfl_xor(r1, 2);
        r0 += __shfl_xor(r0, 4);  r1 += __shfl_xor(r1, 4);
        r0 += __shfl_xor(r0, 8);  r1 += __shfl_xor(r1, 8);
        if (m == j)
            robuf[w][j + 8 * hs] = make_float2(r0, r1);
    }
    __syncthreads();

    if (tid < 16) {
        float o0 = b_out[0] * (float)T;   // reference adds b_out every step
        float o1 = b_out[1] * (float)T;
#pragma unroll
        for (int ww = 0; ww < WAVES; ++ww) {
            float2 pr = robuf[ww][tid];
            o0 += pr.x;  o1 += pr.y;
        }
        int row = blockIdx.x * 16 + tid;
        out[row * 2 + 0] = o0;
        out[row * 2 + 1] = o1;
    }
}

extern "C" void kernel_launch(void* const* d_in, const int* in_sizes, int n_in,
                              void* d_out, int out_size, void* d_ws, size_t ws_size,
                              hipStream_t stream) {
    const float* x     = (const float*)d_in[0];
    const float* W_in  = (const float*)d_in[1];
    const float* b_in  = (const float*)d_in[2];
    const float* ln_g  = (const float*)d_in[3];
    const float* ln_b  = (const float*)d_in[4];
    const float* W_out = (const float*)d_in[5];
    const float* b_out = (const float*)d_in[6];
    float*       out   = (float*)d_out;

    const int T = 512;
    const int B = in_sizes[0] / (T * 4);   // x is [B, T, 4]

    dim3 grid(B / 16);     // one 16-row batch tile per block
    dim3 block(256);       // 8 waves: one N-tile of the hidden dim each
    hipLaunchKernelGGL(snn_fused_kernel, grid, block, 0, stream,
                       x, W_in, b_in, ln_g, ln_b, W_out, b_out, out, B, T);
}
